// RNN_82154134438030
// MI455X (gfx1250) — compile-verified, matrix operand order checked by reading, not run on hardware
//
#include <hip/hip_runtime.h>
#include <math.h>

#define L_SEQ 512
#define BATCH 128
#define DIN   256
#define DH    512
#define SCAN_BLOCKS 32   // 8 M-tiles x 4 N-groups; must all be co-resident

typedef __bf16 bf16;
typedef __attribute__((ext_vector_type(16))) __bf16 v16bf;
typedef __attribute__((ext_vector_type(8)))  float  v8f;
typedef __attribute__((ext_vector_type(4)))  int    v4i;
typedef __attribute__((address_space(1))) v4i gv4i;   // global int4
typedef __attribute__((address_space(3))) v4i lv4i;   // LDS int4

#if defined(__has_builtin)
#if __has_builtin(__builtin_amdgcn_global_load_async_to_lds_b128) && \
    __has_builtin(__builtin_amdgcn_s_wait_asynccnt)
#define USE_ASYNC_LDS 1
#endif
#endif

// Load a 16-element bf16 fragment for one lane.
// Element j maps to k = base + j + (j>=8 ? 8 : 0): two contiguous 16-byte runs
// (ISA 16-bit A/B layout after folding the 8*half lane offset into the pointer).
template <typename P>
__device__ __forceinline__ v16bf load_frag(P p) {
  v16bf f;
#pragma unroll
  for (int j = 0; j < 8; ++j) f[j] = p[j];
#pragma unroll
  for (int j = 0; j < 8; ++j) f[8 + j] = p[16 + j];
  return f;
}

// dst[n*K + k] = (bf16) src[k*N + n]   (convert + transpose weights)
__global__ void cvt_transpose_kernel(const float* __restrict__ src,
                                     bf16* __restrict__ dst, int K, int N) {
  int i = blockIdx.x * blockDim.x + threadIdx.x;
  if (i >= K * N) return;
  int n = i % N;
  int k = i / N;                 // src read is coalesced
  dst[n * K + k] = (bf16)src[i];
}

// dst[i] = (bf16) src[i]
__global__ void cvt_kernel(const float* __restrict__ src, bf16* __restrict__ dst, int n) {
  int i = blockIdx.x * blockDim.x + threadIdx.x;
  if (i < n) dst[i] = (bf16)src[i];
}

__global__ void init_counter_kernel(unsigned int* cnt) { *cnt = 0u; }

// ---------------- Phase 1: xp[m][n] = x[m][:] . Wx[:][n] + bx[n] ----------------
// block: 256 thr = 8 waves; 1 M-tile (16 rows) x 8 N-tiles; grid = 4096*4
__global__ __launch_bounds__(256)
void input_proj_kernel(const float* __restrict__ x, const bf16* __restrict__ WxT,
                       const float* __restrict__ bx, float* __restrict__ xp) {
  __shared__ bf16 aTile[16 * DIN];                    // 8 KB
  const int tid  = threadIdx.x;
  const int lane = tid & 31;
  const int wave = tid >> 5;
  const int mtile = blockIdx.x >> 2;                  // 0..4095
  const int ngrp  = blockIdx.x & 3;                   // 0..3
  const int rowBase = mtile * 16;

  const float* srcA = x + (size_t)rowBase * DIN;
#pragma unroll
  for (int i = 0; i < 16; ++i) {
    int e = i * 256 + tid;
    aTile[e] = (bf16)srcA[e];
  }
  __syncthreads();

  const int ntile = ngrp * 8 + wave;                  // 0..31
  const int col   = ntile * 16 + (lane & 15);
  const int half  = lane >> 4;
  const bf16* aPtr = aTile + (lane & 15) * DIN + 8 * half;
  const bf16* bPtr = WxT + (size_t)col * DIN + 8 * half;

  v8f acc = {};
#pragma unroll
  for (int kk = 0; kk < DIN / 32; ++kk) {             // 8 WMMAs
    v16bf a = load_frag(aPtr + kk * 32);
    v16bf b = load_frag(bPtr + kk * 32);
    acc = __builtin_amdgcn_wmma_f32_16x16x32_bf16(false, a, false, b,
                                                  (short)0, acc, false, false);
  }

  const float bias = bx[col];
  float* outp = xp + (size_t)rowBase * DH + col;
#pragma unroll
  for (int v = 0; v < 8; ++v) {
    size_t off = (size_t)(v + 8 * half) * DH;
    outp[off] = acc[v] + bias;
  }
}

// ---------------- Phase 2: persistent scan over all 512 timesteps ----------------
// grid = exactly 32 blocks; each block owns 1 M-tile x 8 N-tiles for the whole scan.
// Wh column-block lives in VGPRs (128 regs/lane) for the entire kernel.
// h is double-buffered in bf16 (hb0/hb1): produced already-converted in the
// epilogue, consumed next step via async global->LDS DMA (no cvt on the chain).
__global__ __launch_bounds__(256)
void rnn_scan_kernel(const bf16* __restrict__ WhT, const float* __restrict__ bh,
                     float* __restrict__ out, bf16* __restrict__ hb0,
                     bf16* __restrict__ hb1, unsigned int* __restrict__ barrier_cnt) {
  __shared__ bf16 aTile[16 * DH];                     // 16 KB
  const int tid  = threadIdx.x;
  const int lane = tid & 31;
  const int wave = tid >> 5;
  const int mtile = blockIdx.x >> 2;                  // 0..7
  const int ngrp  = blockIdx.x & 3;                   // 0..3
  const int rowBase = mtile * 16;

  const int ntile = ngrp * 8 + wave;
  const int col   = ntile * 16 + (lane & 15);
  const int half  = lane >> 4;
  const bf16* aPtr = aTile + (lane & 15) * DH + 8 * half;
  const bf16* bPtr = WhT + (size_t)col * DH + 8 * half;

  // Hoist this wave's entire Wh column block into registers (16 x 32B = 128 VGPRs).
  v16bf breg[DH / 32];
#pragma unroll
  for (int kk = 0; kk < DH / 32; ++kk) breg[kk] = load_frag(bPtr + kk * 32);
  const float bias = bh[col];

  for (int t = 0; t < L_SEQ; ++t) {
    const bf16* hin  = (t & 1) ? hb1 : hb0;
    bf16*       hout = (t & 1) ? hb0 : hb1;

    // stage 16 x 512 bf16 rows (16 KB, contiguous) of h_prev into LDS
    const bf16* src = hin + (size_t)rowBase * DH;
#ifdef USE_ASYNC_LDS
#pragma unroll
    for (int i = 0; i < 4; ++i) {
      int e = (i * 256 + tid) * 8;                    // 8 bf16 = 16 B per op
      __builtin_amdgcn_global_load_async_to_lds_b128(
          (gv4i*)(src + e), (lv4i*)(aTile + e), 0, 0);
    }
    __builtin_amdgcn_s_wait_asynccnt(0);
#else
    {
      const uint4* s4 = (const uint4*)src;
      uint4* d4 = (uint4*)aTile;
#pragma unroll
      for (int i = 0; i < 4; ++i) d4[i * 256 + tid] = s4[i * 256 + tid];
    }
#endif
    __syncthreads();

    // prefetch xp tile (independent of h) before the WMMA chain
    float* outp = out + (size_t)t * BATCH * DH + (size_t)rowBase * DH + col;
    float xpv[8];
#pragma unroll
    for (int v = 0; v < 8; ++v) xpv[v] = outp[(size_t)(v + 8 * half) * DH];

    v8f acc = {};
#pragma unroll
    for (int kk = 0; kk < DH / 32; ++kk) {            // 16 WMMAs
      v16bf a = load_frag(aPtr + kk * 32);
      acc = __builtin_amdgcn_wmma_f32_16x16x32_bf16(false, a, false, breg[kk],
                                                    (short)0, acc, false, false);
    }

    bf16* houtp = hout + (size_t)rowBase * DH + col;
#pragma unroll
    for (int v = 0; v < 8; ++v) {
      size_t off = (size_t)(v + 8 * half) * DH;
      float hv = tanhf(acc[v] + xpv[v] + bias);
      outp[off]  = hv;                                // fp32 result
      houtp[off] = (bf16)hv;                          // bf16 copy for next step
    }

    if (t != L_SEQ - 1) {
      // ---- software grid barrier (32 co-resident blocks) ----
      __builtin_amdgcn_fence(__ATOMIC_RELEASE, "agent");   // global_wb + drain stores
      __syncthreads();                                     // all waves released
      if (tid == 0) {
        __hip_atomic_fetch_add(barrier_cnt, 1u, __ATOMIC_RELAXED,
                               __HIP_MEMORY_SCOPE_AGENT);
        const unsigned int target = (unsigned int)SCAN_BLOCKS * (unsigned int)(t + 1);
        while (__hip_atomic_load(barrier_cnt, __ATOMIC_RELAXED,
                                 __HIP_MEMORY_SCOPE_AGENT) < target) {
          __builtin_amdgcn_s_sleep(2);
        }
      }
      __syncthreads();
      __builtin_amdgcn_fence(__ATOMIC_ACQUIRE, "agent");   // global_inv: drop stale lines
    }
  }
}

extern "C" void kernel_launch(void* const* d_in, const int* in_sizes, int n_in,
                              void* d_out, int out_size, void* d_ws, size_t ws_size,
                              hipStream_t stream) {
  (void)in_sizes; (void)n_in; (void)out_size; (void)ws_size;
  const float* x  = (const float*)d_in[0];   // [L, B, DIN]
  const float* h0 = (const float*)d_in[1];   // [B, DH]
  const float* Wx = (const float*)d_in[2];   // [DIN, DH]
  const float* bx = (const float*)d_in[3];   // [DH]
  const float* Wh = (const float*)d_in[4];   // [DH, DH]
  const float* bh = (const float*)d_in[5];   // [DH]
  float* out = (float*)d_out;                // [L, B, DH]

  bf16* WxT = (bf16*)d_ws;                           // [DH][DIN]  256 KB
  bf16* WhT = WxT + (size_t)DIN * DH;                // [DH][DH]   512 KB
  bf16* hb0 = WhT + (size_t)DH * DH;                 // [B][DH]    128 KB
  bf16* hb1 = hb0 + (size_t)BATCH * DH;              // [B][DH]    128 KB
  unsigned int* cnt = (unsigned int*)(hb1 + (size_t)BATCH * DH);

  cvt_transpose_kernel<<<(DIN * DH + 255) / 256, 256, 0, stream>>>(Wx, WxT, DIN, DH);
  cvt_transpose_kernel<<<(DH * DH + 255) / 256, 256, 0, stream>>>(Wh, WhT, DH, DH);
  cvt_kernel<<<(BATCH * DH + 255) / 256, 256, 0, stream>>>(h0, hb0, BATCH * DH);
  init_counter_kernel<<<1, 1, 0, stream>>>(cnt);

  // Phase 1: xp for all timesteps, written straight into d_out
  input_proj_kernel<<<(L_SEQ * BATCH / 16) * 4, 256, 0, stream>>>(x, WxT, bx, out);

  // Phase 2: one persistent kernel runs the whole 512-step scan
  rnn_scan_kernel<<<SCAN_BLOCKS, 256, 0, stream>>>(WhT, bh, out, hb0, hb1, cnt);
}
